// MultiHeadAttention_78219944395429
// MI455X (gfx1250) — compile-verified, matrix operand order checked by reading, not run on hardware
//
#include <hip/hip_runtime.h>

// ---------------------------------------------------------------------------
// MHA for MI455X (gfx1250, wave32, WMMA).  B=4, S=2048, D=1024, H=16, DK=64.
// All GEMMs on v_wmma_f32_16x16x32_bf16.  Inputs + weights pre-converted to
// bf16 once (bandwidth-bound, ~4us at 23.3 TB/s) so GEMM hot loops are pure
// {global b128, ds b128, wmma}.  Weight B-tiles staged in LDS with gfx1250
// async-to-LDS copies (ASYNCcnt), double-buffered; the pipeline is unrolled
// by 2 so each stage uses a compile-time LDS buffer (constant ds offsets,
// no accumulator shuffling, no WMMA->VALU hazard bubbles).
// ---------------------------------------------------------------------------

constexpr int B  = 4;
constexpr int S  = 2048;
constexpr int D  = 1024;
constexpr int H  = 16;
constexpr int DK = 64;

typedef __attribute__((ext_vector_type(16))) __bf16          v16bf;
typedef __attribute__((ext_vector_type(8)))  float           v8f;
typedef __attribute__((ext_vector_type(8)))  unsigned short  ushort8;
typedef __attribute__((ext_vector_type(4)))  float           float4v;

union BF16Frag {
    v16bf          v;
    ushort8        h[2];
    unsigned short u[16];
};

// float -> bf16 round-to-nearest-even
__device__ __forceinline__ unsigned short f2bf(float f) {
    unsigned int u = __builtin_bit_cast(unsigned int, f);
    u += 0x7FFFu + ((u >> 16) & 1u);
    return (unsigned short)(u >> 16);
}

// A/B fragment per-lane layout for 16-bit WMMA (16x16x32):
//   lane = m + 16*g ; element e: K = (e>>3)*16 + g*8 + (e&7)
//   => two contiguous 8-element runs at K offsets g*8 and 16+g*8
__device__ __forceinline__ v16bf frag_from_bf16(const unsigned short* rowp, int g) {
    BF16Frag r;
    r.h[0] = *reinterpret_cast<const ushort8*>(rowp + g * 8);
    r.h[1] = *reinterpret_cast<const ushort8*>(rowp + 16 + g * 8);
    return r.v;
}

__device__ __forceinline__ v8f wmma_bf16(v16bf a, v16bf b, v8f c) {
    return __builtin_amdgcn_wmma_f32_16x16x32_bf16(false, a, false, b,
                                                   (short)0, c, false, false);
}

// ---- gfx1250 async global->LDS copy (tracked by ASYNCcnt) ------------------
__device__ __forceinline__ void async_copy_b128(const unsigned short* gsrc,
                                                unsigned short* ldst) {
    unsigned int lds = (unsigned int)(unsigned long long)ldst;  // LDS offset
    asm volatile("global_load_async_to_lds_b128 %0, %1, off"
                 :: "v"(lds), "v"(gsrc) : "memory");
}
__device__ __forceinline__ void wait_async_le2() {
    asm volatile("s_wait_asynccnt 0x2" ::: "memory");
}
__device__ __forceinline__ void wait_async_0() {
    asm volatile("s_wait_asynccnt 0x0" ::: "memory");
}

// ---------------------------------------------------------------------------
// Kernel 0: fp32 -> bf16 conversion (inputs and weights, done once).
// ---------------------------------------------------------------------------
__global__ __launch_bounds__(256) void cvt_bf16_kernel(
    const float* __restrict__ src, unsigned short* __restrict__ dst)
{
    const int i = (blockIdx.x * 256 + threadIdx.x) * 8;
    float4v f0 = *reinterpret_cast<const float4v*>(src + i);
    float4v f1 = *reinterpret_cast<const float4v*>(src + i + 4);
    BF16Frag r;
#pragma unroll
    for (int t = 0; t < 4; ++t) { r.u[t] = f2bf(f0[t]); r.u[4 + t] = f2bf(f1[t]); }
    *reinterpret_cast<ushort8*>(dst + i) = r.h[0];
}

// ---------------------------------------------------------------------------
// Kernel 1: fused QKV projection.  Y = X @ W^T + bias (X, W already bf16).
// Block = 128 (4 waves); wave tile 32x64; block tile 128x64.
// grid = (B*S/128, D/64, 3)
// Head split per reference: o -> h = o % H, dk = o / H.
// Q,K stored [B][H][S][DK] bf16;  V stored transposed [B][H][DK][S] bf16.
// ---------------------------------------------------------------------------
__global__ __launch_bounds__(128) void qkv_proj_kernel(
    const unsigned short* __restrict__ xq, const unsigned short* __restrict__ xk,
    const unsigned short* __restrict__ xv,
    const unsigned short* __restrict__ w_bf,   // [3][D][D] bf16
    const float* __restrict__ b_q, const float* __restrict__ b_k,
    const float* __restrict__ b_v,
    unsigned short* __restrict__ q_ws, unsigned short* __restrict__ k_ws,
    unsigned short* __restrict__ vT_ws)
{
    __shared__ unsigned short Bsh[2][64 * 32];

    const int tid  = threadIdx.x;
    const int wave = tid >> 5;
    const int lane = tid & 31;
    const int m = lane & 15, g = lane >> 4;
    const int z = blockIdx.z;

    const unsigned short* X = (z == 0) ? xq : (z == 1) ? xk : xv;
    const float* bias       = (z == 0) ? b_q : (z == 1) ? b_k : b_v;
    const unsigned short* W = w_bf + (size_t)z * D * D;

    const int n0 = (blockIdx.x * 4 + wave) * 32;   // 32-row wave tile
    const int c0 = blockIdx.y * 64;                // 64-col block tile

    // this thread's two 16B copy chunks of the 64x32 B-tile
    const int cA = 2 * tid,     colA = cA >> 2, qA = (cA & 3) * 8;
    const int cB = 2 * tid + 1, colB = cB >> 2, qB = (cB & 3) * 8;
    const unsigned short* WA = W + (size_t)(c0 + colA) * D + qA;
    const unsigned short* WB = W + (size_t)(c0 + colB) * D + qB;
    unsigned short* LA0 = &Bsh[0][colA * 32 + qA];
    unsigned short* LB0 = &Bsh[0][colB * 32 + qB];
    unsigned short* LA1 = &Bsh[1][colA * 32 + qA];
    unsigned short* LB1 = &Bsh[1][colB * 32 + qB];

    async_copy_b128(WA, LA0);                      // k=0 -> buf0
    async_copy_b128(WB, LB0);

    const unsigned short* arow0 = X + (size_t)(n0 + m) * D;
    const unsigned short* arow1 = X + (size_t)(n0 + 16 + m) * D;

    v8f acc[2][4] = {};

    auto compute = [&](const unsigned short* bsh, int k0) {
        v16bf a0 = frag_from_bf16(arow0 + k0, g);
        v16bf a1 = frag_from_bf16(arow1 + k0, g);
#pragma unroll
        for (int j = 0; j < 4; ++j) {
            v16bf bm = frag_from_bf16(bsh + (j * 16 + m) * 32, g);
            acc[0][j] = wmma_bf16(a0, bm, acc[0][j]);
            acc[1][j] = wmma_bf16(a1, bm, acc[1][j]);
        }
    };

    constexpr int NPAIR = D / 64;                  // 16 double-steps
#pragma unroll 1
    for (int i = 0; i < NPAIR - 1; ++i) {
        const int k0 = i * 64;
        // stage 0: consume buf0 @ k0, prefetch k0+32 -> buf1
        async_copy_b128(WA + k0 + 32, LA1);
        async_copy_b128(WB + k0 + 32, LB1);
        wait_async_le2();
        __syncthreads();
        compute(&Bsh[0][0], k0);
        __syncthreads();
        // stage 1: consume buf1 @ k0+32, prefetch k0+64 -> buf0
        async_copy_b128(WA + k0 + 64, LA0);
        async_copy_b128(WB + k0 + 64, LB0);
        wait_async_le2();
        __syncthreads();
        compute(&Bsh[1][0], k0 + 32);
        __syncthreads();
    }
    {   // tail pair
        const int k0 = (NPAIR - 1) * 64;
        async_copy_b128(WA + k0 + 32, LA1);
        async_copy_b128(WB + k0 + 32, LB1);
        wait_async_le2();
        __syncthreads();
        compute(&Bsh[0][0], k0);
        __syncthreads();
        wait_async_0();
        __syncthreads();
        compute(&Bsh[1][0], k0 + 32);
    }

#pragma unroll
    for (int rt = 0; rt < 2; ++rt) {
#pragma unroll
        for (int j = 0; j < 4; ++j) {
            const int col = c0 + j * 16 + m;       // N = lane&15
            const float bv = bias[col];
            const int hh = col % H;
            const int dk = col / H;
#pragma unroll
            for (int r = 0; r < 8; ++r) {
                const int n  = n0 + rt * 16 + r + 8 * g;
                const int bb = n / S;
                const int s  = n % S;
                const unsigned short hval = f2bf(acc[rt][j][r] + bv);
                if (z == 2) {
                    vT_ws[((size_t)(bb * H + hh) * DK + dk) * S + s] = hval;
                } else {
                    unsigned short* dst = (z == 0) ? q_ws : k_ws;
                    dst[((size_t)(bb * H + hh) * S + s) * DK + dk] = hval;
                }
            }
        }
    }
}

// ---------------------------------------------------------------------------
// Kernel 2: causal flash attention.  One wave per 16-row q-tile.
// grid = (S/64, H, B), block = 128 (4 waves).  Key blocks of 32.
// ---------------------------------------------------------------------------
__global__ __launch_bounds__(128) void attn_kernel(
    const unsigned short* __restrict__ q_ws,
    const unsigned short* __restrict__ k_ws,
    const unsigned short* __restrict__ vT_ws,
    unsigned short* __restrict__ o_ws)
{
    __shared__ unsigned short Pb[4][16 * 32];

    const int wave = threadIdx.x >> 5;
    const int lane = threadIdx.x & 31;
    const int m = lane & 15, g = lane >> 4;
    const int hh = blockIdx.y;
    const int bb = blockIdx.z;
    const int q0 = (blockIdx.x * 4 + wave) * 16;

    const long bh = (long)(bb * H + hh);
    const unsigned short* Qp = q_ws  + bh * S * DK;
    const unsigned short* Kp = k_ws  + bh * S * DK;
    const unsigned short* Vp = vT_ws + bh * DK * S;
    unsigned short* P = &Pb[wave][0];

    v16bf aq[2];
#pragma unroll
    for (int kk = 0; kk < 2; ++kk)
        aq[kk] = frag_from_bf16(Qp + (long)(q0 + m) * DK + kk * 32, g);

    v8f acc[4] = {};
    float mi[8], li[8];
#pragma unroll
    for (int r = 0; r < 8; ++r) { mi[r] = -3.0e38f; li[r] = 0.0f; }

    const int kb_end = (q0 + 15) / 32;
    for (int kb = 0; kb <= kb_end; ++kb) {
        const int kbase = kb * 32;

        // scores: S = Q K^T
        v8f sc[2] = {};
#pragma unroll
        for (int kg = 0; kg < 2; ++kg) {
#pragma unroll
            for (int kk = 0; kk < 2; ++kk) {
                v16bf bk = frag_from_bf16(
                    Kp + (long)(kbase + kg * 16 + m) * DK + kk * 32, g);
                sc[kg] = wmma_bf16(aq[kk], bk, sc[kg]);
            }
        }

        // scale + causal mask
        float sv[2][8], rowmax[8];
#pragma unroll
        for (int r = 0; r < 8; ++r) {
            const int qq = q0 + r + 8 * g;
#pragma unroll
            for (int jj = 0; jj < 2; ++jj) {
                float x = sc[jj][r] * 0.125f;          // 1/sqrt(DK)
                const int kidx = kbase + jj * 16 + m;
                if (kidx > qq) x = -3.0e38f;
                sv[jj][r] = x;
            }
            rowmax[r] = fmaxf(sv[0][r], sv[1][r]);
        }
#pragma unroll
        for (int r = 0; r < 8; ++r)
#pragma unroll
            for (int off = 1; off < 16; off <<= 1)
                rowmax[r] = fmaxf(rowmax[r], __shfl_xor(rowmax[r], off, 32));

        // online softmax update, write P (bf16) to per-wave LDS slab
#pragma unroll
        for (int r = 0; r < 8; ++r) {
            const float mnew = fmaxf(mi[r], rowmax[r]);
            const float corr = __expf(mi[r] - mnew);
            const float p0 = __expf(sv[0][r] - mnew);
            const float p1 = __expf(sv[1][r] - mnew);
            const int row = r + 8 * g;
            P[row * 32 + m]      = f2bf(p0);
            P[row * 32 + 16 + m] = f2bf(p1);
            float rs = p0 + p1;
#pragma unroll
            for (int off = 1; off < 16; off <<= 1)
                rs += __shfl_xor(rs, off, 32);
            li[r] = li[r] * corr + rs;
            mi[r] = mnew;
#pragma unroll
            for (int j = 0; j < 4; ++j) acc[j][r] *= corr;
        }

        // O += P @ V  (A frag re-read from LDS in A layout)
        v16bf ap = frag_from_bf16(P + m * 32, g);
#pragma unroll
        for (int j = 0; j < 4; ++j) {
            v16bf bv = frag_from_bf16(Vp + (long)(j * 16 + m) * S + kbase, g);
            acc[j] = wmma_bf16(ap, bv, acc[j]);
        }
    }

    // normalize; store concat-layout O: feature = h*DK + dk (bf16)
#pragma unroll
    for (int r = 0; r < 8; ++r) {
        const float inv = 1.0f / li[r];
        const int srow = q0 + r + 8 * g;
#pragma unroll
        for (int j = 0; j < 4; ++j) {
            const int dk = j * 16 + m;
            o_ws[(long)(bb * S + srow) * D + hh * DK + dk] = f2bf(acc[j][r] * inv);
        }
    }
}

// ---------------------------------------------------------------------------
// Kernel 3: output projection.  out = O @ Wo^T + b_o (fp32 out).
// Same pipelined structure as kernel 1; pure-bf16 hot loop.
// grid = (B*S/128, D/64)
// ---------------------------------------------------------------------------
__global__ __launch_bounds__(128) void out_proj_kernel(
    const unsigned short* __restrict__ o_ws,
    const unsigned short* __restrict__ wo_bf,
    const float* __restrict__ b_o,
    float* __restrict__ out)
{
    __shared__ unsigned short Bsh[2][64 * 32];

    const int tid  = threadIdx.x;
    const int wave = tid >> 5;
    const int lane = tid & 31;
    const int m = lane & 15, g = lane >> 4;

    const int n0 = (blockIdx.x * 4 + wave) * 32;
    const int c0 = blockIdx.y * 64;

    const int cA = 2 * tid,     colA = cA >> 2, qA = (cA & 3) * 8;
    const int cB = 2 * tid + 1, colB = cB >> 2, qB = (cB & 3) * 8;
    const unsigned short* WA = wo_bf + (size_t)(c0 + colA) * D + qA;
    const unsigned short* WB = wo_bf + (size_t)(c0 + colB) * D + qB;
    unsigned short* LA0 = &Bsh[0][colA * 32 + qA];
    unsigned short* LB0 = &Bsh[0][colB * 32 + qB];
    unsigned short* LA1 = &Bsh[1][colA * 32 + qA];
    unsigned short* LB1 = &Bsh[1][colB * 32 + qB];

    async_copy_b128(WA, LA0);
    async_copy_b128(WB, LB0);

    const unsigned short* arow0 = o_ws + (size_t)(n0 + m) * D;
    const unsigned short* arow1 = o_ws + (size_t)(n0 + 16 + m) * D;

    v8f acc[2][4] = {};

    auto compute = [&](const unsigned short* bsh, int k0) {
        v16bf a0 = frag_from_bf16(arow0 + k0, g);
        v16bf a1 = frag_from_bf16(arow1 + k0, g);
#pragma unroll
        for (int j = 0; j < 4; ++j) {
            v16bf bm = frag_from_bf16(bsh + (j * 16 + m) * 32, g);
            acc[0][j] = wmma_bf16(a0, bm, acc[0][j]);
            acc[1][j] = wmma_bf16(a1, bm, acc[1][j]);
        }
    };

    constexpr int NPAIR = D / 64;
#pragma unroll 1
    for (int i = 0; i < NPAIR - 1; ++i) {
        const int k0 = i * 64;
        async_copy_b128(WA + k0 + 32, LA1);
        async_copy_b128(WB + k0 + 32, LB1);
        wait_async_le2();
        __syncthreads();
        compute(&Bsh[0][0], k0);
        __syncthreads();
        async_copy_b128(WA + k0 + 64, LA0);
        async_copy_b128(WB + k0 + 64, LB0);
        wait_async_le2();
        __syncthreads();
        compute(&Bsh[1][0], k0 + 32);
        __syncthreads();
    }
    {
        const int k0 = (NPAIR - 1) * 64;
        async_copy_b128(WA + k0 + 32, LA1);
        async_copy_b128(WB + k0 + 32, LB1);
        wait_async_le2();
        __syncthreads();
        compute(&Bsh[0][0], k0);
        __syncthreads();
        wait_async_0();
        __syncthreads();
        compute(&Bsh[1][0], k0 + 32);
    }

#pragma unroll
    for (int rt = 0; rt < 2; ++rt) {
#pragma unroll
        for (int j = 0; j < 4; ++j) {
            const int col = c0 + j * 16 + m;
            const float bv = b_o[col];
#pragma unroll
            for (int r = 0; r < 8; ++r) {
                const int n = n0 + rt * 16 + r + 8 * g;
                out[(size_t)n * D + col] = acc[rt][j][r] + bv;
            }
        }
    }
}

// ---------------------------------------------------------------------------
extern "C" void kernel_launch(void* const* d_in, const int* in_sizes, int n_in,
                              void* d_out, int out_size, void* d_ws, size_t ws_size,
                              hipStream_t stream) {
    (void)in_sizes; (void)n_in; (void)out_size; (void)ws_size;
    const float* query = (const float*)d_in[0];
    const float* key_  = (const float*)d_in[1];
    const float* value = (const float*)d_in[2];
    // d_in[3] = causal mask (bool) — statically known, unused
    const float* w_q = (const float*)d_in[4];
    const float* b_q = (const float*)d_in[5];
    const float* w_k = (const float*)d_in[6];
    const float* b_k = (const float*)d_in[7];
    const float* w_v = (const float*)d_in[8];
    const float* b_v = (const float*)d_in[9];
    const float* w_o = (const float*)d_in[10];
    const float* b_o = (const float*)d_in[11];

    const size_t per = (size_t)B * H * S * DK;       // 8M bf16 elements
    unsigned short* q_ws  = (unsigned short*)d_ws;   // [B][H][S][DK]
    unsigned short* k_ws  = q_ws  + per;
    unsigned short* vT_ws = k_ws  + per;             // [B][H][DK][S]
    unsigned short* o_ws  = vT_ws + per;             // [B*S][D]
    unsigned short* xq_bf = o_ws  + per;             // [B*S][D] bf16 inputs
    unsigned short* xk_bf = xq_bf + per;
    unsigned short* xv_bf = xk_bf + per;
    unsigned short* w_bf  = xv_bf + per;             // [4][D][D] bf16

    dim3 blk(128);
    const int cvt_in_blocks = (int)(per / (256 * 8));       // 4096
    const int cvt_w_blocks  = (D * D) / (256 * 8);          // 512

    cvt_bf16_kernel<<<cvt_in_blocks, 256, 0, stream>>>(query, xq_bf);
    cvt_bf16_kernel<<<cvt_in_blocks, 256, 0, stream>>>(key_,  xk_bf);
    cvt_bf16_kernel<<<cvt_in_blocks, 256, 0, stream>>>(value, xv_bf);
    cvt_bf16_kernel<<<cvt_w_blocks, 256, 0, stream>>>(w_q, w_bf + 0 * (size_t)D * D);
    cvt_bf16_kernel<<<cvt_w_blocks, 256, 0, stream>>>(w_k, w_bf + 1 * (size_t)D * D);
    cvt_bf16_kernel<<<cvt_w_blocks, 256, 0, stream>>>(w_v, w_bf + 2 * (size_t)D * D);
    cvt_bf16_kernel<<<cvt_w_blocks, 256, 0, stream>>>(w_o, w_bf + 3 * (size_t)D * D);

    dim3 g1(B * S / 128, D / 64, 3);
    qkv_proj_kernel<<<g1, blk, 0, stream>>>(xq_bf, xk_bf, xv_bf, w_bf,
                                            b_q, b_k, b_v, q_ws, k_ws, vT_ws);

    dim3 g2(S / 64, H, B);
    attn_kernel<<<g2, blk, 0, stream>>>(q_ws, k_ws, vT_ws, o_ws);

    dim3 g3(B * S / 128, D / 64);
    out_proj_kernel<<<g3, blk, 0, stream>>>(o_ws, w_bf + 3 * (size_t)D * D,
                                            b_o, (float*)d_out);
}
